// Ladder_16681652977739
// MI455X (gfx1250) — matrix operand order repaired and checked by reading, not compiled, not run on hardware
//
#include <hip/hip_runtime.h>
#include <stdint.h>

// ---------------------------------------------------------------------------
// CDNA5 (gfx1250) implementation of the autoregressive Ising sampler.
// All dense GEMMs run on v_wmma_f32_16x16x32_bf16 (wave32, 16x16 tiles).
// ---------------------------------------------------------------------------

typedef __attribute__((ext_vector_type(16))) __bf16 bf16x16;
typedef __attribute__((ext_vector_type(8)))  float  f32x8;
typedef unsigned short u16;

#define B_   256
#define L_   16
#define C_   512
#define Q_   2
#define FS_  15
#define G4C  2048   // 4*C

__device__ __forceinline__ u16 f2bf(float f) {
    unsigned u = __float_as_uint(f);
    u += 0x7FFFu + ((u >> 16) & 1u);       // round-to-nearest-even
    return (u16)(u >> 16);
}

__device__ __forceinline__ float sigm(float x) {
    return 1.0f / (1.0f + __expf(-x));
}

__device__ __forceinline__ float u01hash(unsigned step, unsigned b) {
    unsigned x = step * 0x9E3779B1u ^ (b + 1u) * 0x85EBCA77u;
    x ^= x >> 16; x *= 0x7FEB352Du;
    x ^= x >> 15; x *= 0x846CA68Bu;
    x ^= x >> 16;
    return (float)(x >> 8) * (1.0f / 16777216.0f);
}

union FragBF { bf16x16 v; uint4 q[2]; };

// A-matrix 16x32 bf16 fragment (ISA 7.12.2): lane m=lane&15, K-half by lane>=16.
__device__ __forceinline__ bf16x16 load_a_frag(const u16* __restrict__ X, int ld,
                                               int mtile, int k0, int lane) {
    int m  = mtile * 16 + (lane & 15);
    int kb = (lane >> 4) * 8;
    const u16* p = X + m * ld + k0 + kb;
    FragBF f;
    f.q[0] = *(const uint4*)(p);        // elements 0..7  -> K = k0+kb .. +7
    f.q[1] = *(const uint4*)(p + 16);   // elements 8..15 -> K = k0+16+kb .. +7
    return f.v;
}

// B-matrix 32x16 bf16 fragment. W is [N,K] row-major, so B[k][n] = W[n][k]:
// each lane reads K = k0+kb..+7 and k0+16+kb..+7 from one weight row.
__device__ __forceinline__ bf16x16 load_b_frag(const u16* __restrict__ W, int ld,
                                               int ntile, int k0, int lane) {
    int n  = ntile * 16 + (lane & 15);
    int kb = (lane >> 4) * 8;
    const u16* p = W + n * ld + k0 + kb;
    FragBF f;
    f.q[0] = *(const uint4*)(p);
    f.q[1] = *(const uint4*)(p + 16);
    return f.v;
}

// ------------------------------- utility kernels ---------------------------

__global__ void k_cvt_bf16(const float* __restrict__ s, u16* __restrict__ d, int n) {
    for (int i = blockIdx.x * blockDim.x + threadIdx.x; i < n; i += gridDim.x * blockDim.x)
        d[i] = f2bf(s[i]);
}

// w_ih0 is [2048,16]; pad K to 32 so the WMMA K-loop stays uniform.
__global__ void k_pad_ih0(const float* __restrict__ src, u16* __restrict__ dst) {
    int i = blockIdx.x * blockDim.x + threadIdx.x;    // 2048*32
    if (i >= 2048 * 32) return;
    int n = i >> 5, k = i & 31;
    dst[i] = (k < 16) ? f2bf(src[n * 16 + k]) : (u16)0;
}

// ------------------------------- MLP GEMM (WMMA) ---------------------------
// Z[256,512] = relu( Xbf[256,512] * W1bf[512,512]^T + b1 ),  fp32 out.
__global__ __launch_bounds__(256) void k_gemm_relu(const u16* __restrict__ X,
                                                   const u16* __restrict__ W,
                                                   const float* __restrict__ bias,
                                                   float* __restrict__ Z) {
    int lane  = threadIdx.x & 31;
    int wid   = (blockIdx.x * blockDim.x + threadIdx.x) >> 5;  // 0..511
    int mtile = wid >> 5;   // 0..15
    int ntile = wid & 31;   // 0..31
    f32x8 acc = {};
#pragma unroll 4
    for (int k0 = 0; k0 < 512; k0 += 32) {
        if (k0 + 64 < 512)
            __builtin_prefetch(W + (ntile * 16 + (lane & 15)) * 512 + k0 + 64, 0, 1);
        bf16x16 a = load_a_frag(X, 512, mtile, k0, lane);
        bf16x16 b = load_b_frag(W, 512, ntile, k0, lane);
        acc = __builtin_amdgcn_wmma_f32_16x16x32_bf16(false, a, false, b,
                                                      (short)0, acc, false, false);
    }
    int n   = ntile * 16 + (lane & 15);
    int mhi = (lane >> 4) * 8;
    float bn = bias[n];
#pragma unroll
    for (int r = 0; r < 8; ++r) {
        float v = acc[r] + bn;
        Z[(mtile * 16 + mhi + r) * 512 + n] = v > 0.f ? v : 0.f;
    }
}

// ------------------------------- LSTM gate GEMM (WMMA) ---------------------
// G[256,2048] = A1[256,K1]*W1^T + A2[256,512]*W2^T + bi + bh   (fp32 out)
__global__ __launch_bounds__(256) void k_gates(const u16* __restrict__ A1, int K1,
                                               const u16* __restrict__ W1,
                                               const u16* __restrict__ A2,
                                               const u16* __restrict__ W2,
                                               const float* __restrict__ bi,
                                               const float* __restrict__ bh,
                                               float* __restrict__ G) {
    int lane  = threadIdx.x & 31;
    int wid   = (blockIdx.x * blockDim.x + threadIdx.x) >> 5;  // 0..2047
    int mtile = wid >> 7;    // 0..15
    int ntile = wid & 127;   // 0..127
    f32x8 acc = {};
    for (int k0 = 0; k0 < K1; k0 += 32) {
        bf16x16 a = load_a_frag(A1, K1, mtile, k0, lane);
        bf16x16 b = load_b_frag(W1, K1, ntile, k0, lane);
        acc = __builtin_amdgcn_wmma_f32_16x16x32_bf16(false, a, false, b,
                                                      (short)0, acc, false, false);
    }
#pragma unroll 4
    for (int k0 = 0; k0 < 512; k0 += 32) {
        bf16x16 a = load_a_frag(A2, 512, mtile, k0, lane);
        bf16x16 b = load_b_frag(W2, 512, ntile, k0, lane);
        acc = __builtin_amdgcn_wmma_f32_16x16x32_bf16(false, a, false, b,
                                                      (short)0, acc, false, false);
    }
    int n   = ntile * 16 + (lane & 15);
    int mhi = (lane >> 4) * 8;
    float bn = bi[n] + bh[n];
#pragma unroll
    for (int r = 0; r < 8; ++r)
        G[(mtile * 16 + mhi + r) * G4C + n] = acc[r] + bn;
}

// LSTM elementwise: torch gate order i,f,g,o in chunks of 512.
__global__ void k_lstm_elem(const float* __restrict__ G, float* __restrict__ c,
                            float* __restrict__ h, u16* __restrict__ hbf) {
    int i = blockIdx.x * blockDim.x + threadIdx.x;    // < 256*512
    if (i >= B_ * C_) return;
    int b = i >> 9, ch = i & 511;
    const float* g = G + b * G4C;
    float gi = g[ch], gf = g[512 + ch], gg = g[1024 + ch], go = g[1536 + ch];
    float cn = sigm(gf) * c[i] + sigm(gi) * tanhf(gg);
    float hn = sigm(go) * tanhf(cn);
    c[i] = cn; h[i] = hn; hbf[i] = f2bf(hn);
}

// Build LSTM layer-0 input: u[:, srow:srow+4, scol:scol+4] -> [B,32] bf16 (K padded).
__global__ void k_build_x0(const float* __restrict__ uw, u16* __restrict__ x0,
                           int srow, int scol) {
    int i = blockIdx.x * blockDim.x + threadIdx.x;    // < 256*32
    if (i >= B_ * 32) return;
    int b = i >> 5, e = i & 31;
    float v = 0.f;
    if (e < 16) v = uw[b * 256 + (srow + (e >> 2)) * 16 + (scol + (e & 3))];
    x0[i] = f2bf(v);
}

// Embedding gather + hidden-state sum -> bf16 MLP input.
__global__ void k_gather(const int* __restrict__ vw, const float* __restrict__ emb,
                         const float* __restrict__ hj, const float* __restrict__ hi,
                         u16* __restrict__ xbf, int arow, int bcol) {
    __shared__ int sidx[FS_];
    int b = blockIdx.x;
    int t = threadIdx.x;
    if (t < FS_) {
        int r = t >> 2, cc = t & 3;   // patch row-major; element 15 (target) excluded
        sidx[t] = vw[b * 256 + (arow + r) * 16 + (bcol + cc)];
    }
    __syncthreads();
    for (int c = t; c < C_; c += blockDim.x) {
        float acc = hj[b * C_ + c] + hi[b * C_ + c];
#pragma unroll
        for (int f = 0; f < FS_; ++f)
            acc += emb[(f * Q_ + sidx[f]) * C_ + c];
        xbf[b * C_ + c] = f2bf(acc);
    }
}

// Head: logits = Z @ w2^T + b2 (Q=2), log_softmax, scatter, sample, update v/u.
__global__ void k_head(const float* __restrict__ Z, const float* __restrict__ w2,
                       const float* __restrict__ b2, float* __restrict__ softout,
                       float* __restrict__ uw, int* __restrict__ vw,
                       int pr, int pc, unsigned step) {
    int b = threadIdx.x;
    if (b >= B_) return;
    const float* z = Z + b * C_;
    float d0 = b2[0], d1 = b2[1];
    for (int k = 0; k < C_; ++k) {
        float zz = z[k];
        d0 = fmaf(zz, w2[k], d0);
        d1 = fmaf(zz, w2[C_ + k], d1);
    }
    float m   = fmaxf(d0, d1);
    float lse = m + __logf(__expf(d0 - m) + __expf(d1 - m));
    float l0 = d0 - lse, l1 = d1 - lse;
    int o = (b * 256 + pr * 16 + pc) * Q_;
    softout[o]     = l0;
    softout[o + 1] = l1;
    int s = (u01hash(step, (unsigned)b) < __expf(l1)) ? 1 : 0;
    vw[b * 256 + pr * 16 + pc] = s;
    uw[b * 256 + pr * 16 + pc] = (float)s;
}

__global__ void k_finalize(const float* __restrict__ uw, const int* __restrict__ vw,
                           float* __restrict__ ou, float* __restrict__ ov) {
    int i = blockIdx.x * blockDim.x + threadIdx.x;
    if (i >= B_ * L_ * L_) return;
    ou[i] = uw[i];
    ov[i] = (float)vw[i];
}

// --------------------------------- host orchestration ----------------------

extern "C" void kernel_launch(void* const* d_in, const int* in_sizes, int n_in,
                              void* d_out, int out_size, void* d_ws, size_t ws_size,
                              hipStream_t stream) {
    (void)in_sizes; (void)n_in; (void)out_size;

    const float* u_in  = (const float*)d_in[0];
    const int*   v_in  = (const int*)  d_in[1];
    const float* emb   = (const float*)d_in[2];
    const float* w1    = (const float*)d_in[3];
    const float* b1    = (const float*)d_in[4];
    const float* w2    = (const float*)d_in[5];
    const float* b2    = (const float*)d_in[6];
    const float* w_ih0 = (const float*)d_in[7];
    const float* w_hh0 = (const float*)d_in[8];
    const float* b_ih0 = (const float*)d_in[9];
    const float* b_hh0 = (const float*)d_in[10];
    const float* w_ih1 = (const float*)d_in[11];
    const float* w_hh1 = (const float*)d_in[12];
    const float* b_ih1 = (const float*)d_in[13];
    const float* b_hh1 = (const float*)d_in[14];

    float* out_soft = (float*)d_out;               // [B,L,L,Q]
    float* out_u    = out_soft + B_ * L_ * L_ * Q_;
    float* out_v    = out_u + B_ * L_ * L_;

    // workspace carve-out (256B aligned)
    char*  base = (char*)d_ws;
    size_t off  = 0;
    auto alloc = [&](size_t bytes) -> char* {
        char* p = base + off;
        off = (off + bytes + 255) & ~(size_t)255;
        return p;
    };
    const int HC = B_ * C_;                         // 131072 elems per (sector,layer)
    u16*   w1b   = (u16*)  alloc((size_t)512 * 512 * 2);
    u16*   wih0b = (u16*)  alloc((size_t)2048 * 32 * 2);
    u16*   whh0b = (u16*)  alloc((size_t)2048 * 512 * 2);
    u16*   wih1b = (u16*)  alloc((size_t)2048 * 512 * 2);
    u16*   whh1b = (u16*)  alloc((size_t)2048 * 512 * 2);
    u16*   xbf   = (u16*)  alloc((size_t)HC * 2);
    float* zf    = (float*)alloc((size_t)HC * 4);
    u16*   x0bf  = (u16*)  alloc((size_t)B_ * 32 * 2);
    float* gates = (float*)alloc((size_t)B_ * G4C * 4);
    float* hj    = (float*)alloc((size_t)8 * HC * 4);   // memj[4] x layers[2] h (f32)
    float* cjs   = (float*)alloc((size_t)8 * HC * 4);
    u16*   hjb   = (u16*)  alloc((size_t)8 * HC * 2);
    float* hi    = (float*)alloc((size_t)8 * HC * 4);   // memi[4] x layers[2]
    float* cis   = (float*)alloc((size_t)8 * HC * 4);
    u16*   hib   = (u16*)  alloc((size_t)8 * HC * 2);
    int*   vw    = (int*)  alloc((size_t)B_ * 256 * 4);
    float* uw    = (float*)alloc((size_t)B_ * 256 * 4);
    if (off > ws_size) return;

    // per-call init (deterministic)
    hipMemsetAsync(out_soft, 0, (size_t)B_ * L_ * L_ * Q_ * 4, stream);
    hipMemsetAsync(hj, 0, (size_t)8 * HC * 4, stream);
    hipMemsetAsync(cjs, 0, (size_t)8 * HC * 4, stream);
    hipMemsetAsync(hjb, 0, (size_t)8 * HC * 2, stream);
    hipMemsetAsync(hi, 0, (size_t)8 * HC * 4, stream);
    hipMemsetAsync(cis, 0, (size_t)8 * HC * 4, stream);
    hipMemsetAsync(hib, 0, (size_t)8 * HC * 2, stream);
    hipMemcpyAsync(vw, v_in, (size_t)B_ * 256 * 4, hipMemcpyDeviceToDevice, stream);
    hipMemcpyAsync(uw, u_in, (size_t)B_ * 256 * 4, hipMemcpyDeviceToDevice, stream);

    k_cvt_bf16<<<512, 256, 0, stream>>>(w1, w1b, 512 * 512);
    k_pad_ih0 <<<256, 256, 0, stream>>>(w_ih0, wih0b);
    k_cvt_bf16<<<2048, 256, 0, stream>>>(w_hh0, whh0b, 2048 * 512);
    k_cvt_bf16<<<2048, 256, 0, stream>>>(w_ih1, wih1b, 2048 * 512);
    k_cvt_bf16<<<2048, 256, 0, stream>>>(w_hh1, whh1b, 2048 * 512);

    auto lstm_step = [&](int s, float* Hst, float* Cst, u16* Hb, int srow, int scol) {
        float* h0 = Hst + (size_t)(s * 2 + 0) * HC;
        float* c0 = Cst + (size_t)(s * 2 + 0) * HC;
        u16*   b0 = Hb  + (size_t)(s * 2 + 0) * HC;
        float* h1 = Hst + (size_t)(s * 2 + 1) * HC;
        float* c1 = Cst + (size_t)(s * 2 + 1) * HC;
        u16*   b1v = Hb + (size_t)(s * 2 + 1) * HC;
        k_build_x0<<<32, 256, 0, stream>>>(uw, x0bf, srow, scol);
        k_gates<<<256, 256, 0, stream>>>(x0bf, 32, wih0b, b0, whh0b, b_ih0, b_hh0, gates);
        k_lstm_elem<<<512, 256, 0, stream>>>(gates, c0, h0, b0);
        k_gates<<<256, 256, 0, stream>>>(b0, 512, wih1b, b1v, whh1b, b_ih1, b_hh1, gates);
        k_lstm_elem<<<512, 256, 0, stream>>>(gates, c1, h1, b1v);
    };

    int sector0 = 0, sector1 = 0;
    unsigned step = 0;
    for (int i = 0; i < 3; ++i) {
        for (int j = 0; j < 3; ++j) {
            int si = i * 4, sj = j * 4;
            if (sj != sector1) {
                sector0 = si; sector1 = sj;
                for (int s = 0; s < 4; ++s)
                    lstm_step(s, hj, cjs, hjb, s * 4, sector1);
            }
            if (si != sector0) {             // provably dead for this geometry; kept faithful
                sector0 = si;
                lstm_step(sector0 / 4, hi, cis, hib, sector0 - 4, sector1);
            }
            const float* h0j = hj + (size_t)(i * 2) * HC;   // memj[i] layer-0 h
            const float* h0i = hi + (size_t)(j * 2) * HC;   // memi[j] layer-0 h
            for (int ii = 0; ii < 4; ++ii) {
                for (int jj = 0; jj < 4; ++jj) {
                    int a = si + ii + 1, bcol = sj + jj + 1;
                    int pr = a + 3, pc = bcol + 3;
                    k_gather<<<B_, 256, 0, stream>>>(vw, emb, h0j, h0i, xbf, a, bcol);
                    k_gemm_relu<<<64, 256, 0, stream>>>(xbf, w1b, b1, zf);
                    k_head<<<1, 256, 0, stream>>>(zf, w2, b2, out_soft, uw, vw, pr, pc, step);
                    ++step;
                }
            }
        }
    }
    k_finalize<<<256, 256, 0, stream>>>(uw, vw, out_u, out_v);
}
